// SpUNetBase_64785286693641
// MI455X (gfx1250) — matrix operand order repaired and checked by reading, not compiled, not run on hardware
//
#include <hip/hip_runtime.h>
#include <cstdio>
#include <cstring>
#include <algorithm>

typedef __attribute__((ext_vector_type(16))) _Float16 v16h;
typedef __attribute__((ext_vector_type(8)))  _Float16 v8h;
typedef __attribute__((ext_vector_type(8)))  float    v8f;

#define EPSBN 1e-3f

// ---------------------------------------------------------------------------
// WMMA gather-GEMM sparse conv:  out[i,:] = sum_k f16feat[idx[i,k], :] @ W16[k]
// A: [padRow+1, CinP] f16 (row padRow all zeros). idx==null -> identity rows.
// W16: [K, CinP, Cout] f16 (rows c>=Cin zero-padded). out: [Nout, Cout] f32.
// Block = 256 threads = 8 waves; tile 128(M) x 64(N); wave grid 4x2, each wave
// computes 32x32 via four v_wmma_f32_16x16x32_f16. LDS double-buffered:
// one barrier per 32-wide K-chunk, next chunk's global loads overlap WMMAs.
// ---------------------------------------------------------------------------
__launch_bounds__(256)
__global__ void k_sconv(const _Float16* __restrict__ A, const int* __restrict__ idx,
                        const _Float16* __restrict__ W, float* __restrict__ out,
                        int Nout, int K, int CinP, int Cout, int padRow)
{
  __shared__ __align__(16) _Float16 As[2][128][40];   // 20.5 KB  [m][k-chunk]
  __shared__ __align__(16) _Float16 Bs[2][64][40];    // 10.3 KB  [n][k-chunk]
  const int tid = threadIdx.x;
  const int mtiles = (Nout + 127) >> 7;
  const int bm = blockIdx.x % mtiles;
  const int bn = blockIdx.x / mtiles;
  const int m0 = bm << 7;
  const int n0 = bn << 6;

  const int lane = tid & 31;
  const int wave = tid >> 5;
  const int wx = wave & 3;        // 4 wave-rows of 32
  const int wy = wave >> 2;       // 2 wave-cols of 32
  const int l16 = lane & 15;
  const bool hi = lane >= 16;
  const int mw = wx << 5;
  const int nw = wy << 5;

  v8f acc00 = {}, acc01 = {}, acc10 = {}, acc11 = {};

  // staging assignments
  const int am   = tid >> 1;          // A row 0..127
  const int ap16 = (tid & 1) << 4;    // half offset 0/16 (16 halves per thread)
  const int bnn  = tid & 63;          // B col 0..63
  const int bc8  = (tid >> 6) << 3;   // c offset 0/8/16/24
  const int gcol = n0 + bnn;
  const int gr   = m0 + am;

  auto stage = [&](int buf, int k, int c0) {
    int r;
    if (idx) r = (gr < Nout) ? idx[(size_t)gr * K + k] : padRow;
    else     r = (gr < Nout) ? gr : padRow;
    const _Float16* __restrict__ arow = A + (size_t)r * CinP + c0 + ap16;
    *(v8h*)&As[buf][am][ap16]     = *(const v8h*)(arow);
    *(v8h*)&As[buf][am][ap16 + 8] = *(const v8h*)(arow + 8);
    v8h bv = {};
    if (gcol < Cout) {
      const _Float16* __restrict__ wp = W + ((size_t)k * CinP + c0 + bc8) * Cout + gcol;
      #pragma unroll
      for (int j = 0; j < 8; ++j) bv[j] = wp[(size_t)j * Cout];
    }
    *(v8h*)&Bs[buf][bnn][bc8] = bv;
  };

  const int nc = CinP >> 5;
  const int total = K * nc;

  stage(0, 0, 0);
  __syncthreads();

  int kc = 0, c0c = 0;                // coords of chunk just staged
  for (int cc = 0; cc < total; ++cc) {
    const int cur = cc & 1;
    c0c += 32; if (c0c >= CinP) { c0c = 0; ++kc; }
    if (cc + 1 < total) stage(cur ^ 1, kc, c0c);

    // A fragments: lanes 0-15 K{0..7,16..23}, lanes 16-31 K{8..15,24..31}
    const _Float16* ap0 = &As[cur][mw + l16][hi ? 8 : 0];
    v8h a0l = *(const v8h*)ap0;
    v8h a0h = *(const v8h*)(ap0 + 16);
    const _Float16* ap1 = &As[cur][mw + 16 + l16][hi ? 8 : 0];
    v8h a1l = *(const v8h*)ap1;
    v8h a1h = *(const v8h*)(ap1 + 16);
    // B fragments: lanes 0-15 col=l16 K{0..7,16..23}; lanes 16-31 col=l16 K{8..15,24..31}
    const _Float16* bp0 = &Bs[cur][nw + l16][hi ? 8 : 0];
    v8h b0l = *(const v8h*)bp0;
    v8h b0h = *(const v8h*)(bp0 + 16);
    const _Float16* bp1 = &Bs[cur][nw + 16 + l16][hi ? 8 : 0];
    v8h b1l = *(const v8h*)bp1;
    v8h b1h = *(const v8h*)(bp1 + 16);

    v16h af0, af1, bf0, bf1;
    #pragma unroll
    for (int j = 0; j < 8; ++j) {
      af0[j] = a0l[j]; af0[j + 8] = a0h[j];
      af1[j] = a1l[j]; af1[j + 8] = a1h[j];
      bf0[j] = b0l[j]; bf0[j + 8] = b0h[j];
      bf1[j] = b1l[j]; bf1[j + 8] = b1h[j];
    }
    acc00 = __builtin_amdgcn_wmma_f32_16x16x32_f16(false, af0, false, bf0, (short)0, acc00, false, false);
    acc01 = __builtin_amdgcn_wmma_f32_16x16x32_f16(false, af0, false, bf1, (short)0, acc01, false, false);
    acc10 = __builtin_amdgcn_wmma_f32_16x16x32_f16(false, af1, false, bf0, (short)0, acc10, false, false);
    acc11 = __builtin_amdgcn_wmma_f32_16x16x32_f16(false, af1, false, bf1, (short)0, acc11, false, false);
    __syncthreads();
  }

  // C/D layout: VGPR j -> M=j (lanes 0-15) / M=j+8 (lanes 16-31), N=l16
  const int col0 = n0 + nw + l16;
  const int col1 = col0 + 16;
  const int rb = m0 + mw + (hi ? 8 : 0);
  #pragma unroll
  for (int j = 0; j < 8; ++j) {
    int r0 = rb + j;
    int r1 = rb + 16 + j;
    if (col0 < Cout) {
      if (r0 < Nout) out[(size_t)r0 * Cout + col0] = acc00[j];
      if (r1 < Nout) out[(size_t)r1 * Cout + col0] = acc10[j];
    }
    if (col1 < Cout) {
      if (r0 < Nout) out[(size_t)r0 * Cout + col1] = acc01[j];
      if (r1 < Nout) out[(size_t)r1 * Cout + col1] = acc11[j];
    }
  }
}

// ---------------------------------------------------------------------------
// Per-channel sum / sum-of-squares. One block per 32 columns (coalesced).
// stats[c] = sum, stats[C+c] = sumsq. Requires C % 32 == 0.
// ---------------------------------------------------------------------------
__launch_bounds__(256)
__global__ void k_stats(const float* __restrict__ x, float* __restrict__ stats, int N, int C)
{
  const int cl = threadIdx.x & 31;
  const int col = (blockIdx.x << 5) + cl;
  const int rl = threadIdx.x >> 5;   // 0..7
  float s = 0.f, q = 0.f;
  for (int r = rl; r < N; r += 8) {
    float v = x[(size_t)r * C + col];
    s += v; q += v * v;
  }
  __shared__ float sh[256], qh[256];
  sh[threadIdx.x] = s; qh[threadIdx.x] = q;
  __syncthreads();
  if (rl == 0) {
    #pragma unroll
    for (int j = 1; j < 8; ++j) { s += sh[(j << 5) + cl]; q += qh[(j << 5) + cl]; }
    stats[col] = s; stats[C + col] = q;
  }
}

// ---------------------------------------------------------------------------
// y = g*(x-mean)*rsqrt(var+eps)+b  [+res] [relu] -> dst32 (strided) + dst16
// ---------------------------------------------------------------------------
__launch_bounds__(256)
__global__ void k_bn_act(const float* __restrict__ t, const float* __restrict__ stats,
                         const float* __restrict__ g, const float* __restrict__ b,
                         const float* __restrict__ res,
                         float* __restrict__ dst32, int s32, int o32,
                         _Float16* __restrict__ dst16, int s16, int o16,
                         int N, int C, int relu)
{
  size_t i = (size_t)blockIdx.x * 256 + threadIdx.x;
  size_t tot = (size_t)N * C;
  if (i >= tot) return;
  int row = (int)(i / C), c = (int)(i % C);
  float invN = 1.0f / (float)N;
  float m = stats[c] * invN;
  float v = fmaxf(stats[C + c] * invN - m * m, 0.0f);
  float y = g[c] * (t[i] - m) * rsqrtf(v + EPSBN) + b[c];
  if (res) y += res[(size_t)row * C + c];
  if (relu) y = fmaxf(y, 0.0f);
  if (dst32) dst32[(size_t)row * s32 + o32 + c] = y;
  if (dst16) dst16[(size_t)row * s16 + o16 + c] = (_Float16)y;
}

// fp32 [N, srcStride] -> strided fp32 and/or f16 copies
__launch_bounds__(256)
__global__ void k_pack(const float* __restrict__ src, int srcStride,
                       float* __restrict__ dst32, int s32, int o32,
                       _Float16* __restrict__ dst16, int s16, int o16,
                       int N, int C)
{
  size_t i = (size_t)blockIdx.x * 256 + threadIdx.x;
  if (i >= (size_t)N * C) return;
  int row = (int)(i / C), c = (int)(i % C);
  float v = src[(size_t)row * srcStride + c];
  if (dst32) dst32[(size_t)row * s32 + o32 + c] = v;
  if (dst16) dst16[(size_t)row * s16 + o16 + c] = (_Float16)v;
}

// W fp32 [K,Cin,Cout] -> f16 [K,CinP,Cout], rows c>=Cin zeroed
__launch_bounds__(256)
__global__ void k_castw(const float* __restrict__ src, _Float16* __restrict__ dst,
                        int K, int Cin, int CinP, int Cout)
{
  size_t i = (size_t)blockIdx.x * 256 + threadIdx.x;
  size_t tot = (size_t)K * CinP * Cout;
  if (i >= tot) return;
  int n = (int)(i % Cout);
  int c = (int)((i / Cout) % CinP);
  int k = (int)(i / ((size_t)Cout * CinP));
  dst[i] = (c < Cin) ? (_Float16)src[((size_t)k * Cin + c) * Cout + n] : (_Float16)0.0f;
}

// inverse-conv rulebook: idx8[i][d] = (off[i]==d) ? par[i] : pad
__launch_bounds__(256)
__global__ void k_invidx(const int* __restrict__ par, const int* __restrict__ off,
                         int* __restrict__ idx8, int N, int pad)
{
  int i = blockIdx.x * 256 + threadIdx.x;
  if (i >= N) return;
  int p = par[i], o = off[i];
  #pragma unroll
  for (int d = 0; d < 8; ++d) idx8[(size_t)i * 8 + d] = (d == o) ? p : pad;
}

// final linear: out = h[N,96] @ W[96,13] + b
__launch_bounds__(128)
__global__ void k_final(const float* __restrict__ h, const float* __restrict__ W,
                        const float* __restrict__ bias, float* __restrict__ out, int N)
{
  __shared__ float Ws[96 * 13];
  for (int j = threadIdx.x; j < 96 * 13; j += 128) Ws[j] = W[j];
  __syncthreads();
  int r = blockIdx.x * 128 + threadIdx.x;
  if (r >= N) return;
  const float* hr = h + (size_t)r * 96;
  float acc[13];
  #pragma unroll
  for (int c = 0; c < 13; ++c) acc[c] = bias[c];
  for (int k = 0; k < 96; ++k) {
    float hv = hr[k];
    #pragma unroll
    for (int c = 0; c < 13; ++c) acc[c] += hv * Ws[k * 13 + c];
  }
  #pragma unroll
  for (int c = 0; c < 13; ++c) out[(size_t)r * 13 + c] = acc[c];
}

// ===========================================================================
extern "C" void kernel_launch(void* const* d_in, const int* in_sizes, int n_in,
                              void* d_out, int out_size, void* d_ws, size_t ws_size,
                              hipStream_t stream)
{
  (void)out_size; (void)ws_size;
  const float* x = (const float*)d_in[0];
  const int* nb5 = (const int*)d_in[1];
  const int* nb3[5]; for (int l = 0; l < 5; ++l) nb3[l] = (const int*)d_in[2 + l];
  const int* dwn[4]; const int* par[4]; const int* off[4];
  for (int s = 0; s < 4; ++s) {
    dwn[s] = (const int*)d_in[7 + 3 * s];
    par[s] = (const int*)d_in[8 + 3 * s];
    off[s] = (const int*)d_in[9 + 3 * s];
  }
  int Nl[5];
  Nl[0] = in_sizes[8]; Nl[1] = in_sizes[11]; Nl[2] = in_sizes[14];
  Nl[3] = in_sizes[17]; Nl[4] = in_sizes[6] / 27;

  const int CH[8] = {32, 64, 128, 256, 256, 128, 96, 96};
  const int LY[8] = {2, 3, 4, 6, 2, 2, 2, 2};

  // ---- parameter table in _make_params insertion order ----
  struct Param { char name[48]; size_t cnt; const float* p; };
  Param tab[220]; int np = 0;
  auto addp = [&](const char* s, size_t cnt) {
    strncpy(tab[np].name, s, 47); tab[np].name[47] = 0;
    tab[np].cnt = cnt; tab[np].p = nullptr; np++;
  };
  char nmb[64];
  addp("conv_input.w", (size_t)125 * 4 * 32);
  addp("conv_input.bn.g", 32); addp("conv_input.bn.b", 32);
  {
    int enc = 32, dec = 96;
    for (int s = 0; s < 4; ++s) {
      snprintf(nmb, sizeof nmb, "down%d.w", s);   addp(nmb, (size_t)8 * enc * CH[s]);
      snprintf(nmb, sizeof nmb, "down%d.bn.g", s); addp(nmb, CH[s]);
      snprintf(nmb, sizeof nmb, "down%d.bn.b", s); addp(nmb, CH[s]);
      for (int i = 0; i < LY[s]; ++i) {
        int c = CH[s];
        snprintf(nmb, sizeof nmb, "enc%d.%d.conv1.w", s, i); addp(nmb, (size_t)27 * c * c);
        snprintf(nmb, sizeof nmb, "enc%d.%d.bn1.g", s, i);   addp(nmb, c);
        snprintf(nmb, sizeof nmb, "enc%d.%d.bn1.b", s, i);   addp(nmb, c);
        snprintf(nmb, sizeof nmb, "enc%d.%d.conv2.w", s, i); addp(nmb, (size_t)27 * c * c);
        snprintf(nmb, sizeof nmb, "enc%d.%d.bn2.g", s, i);   addp(nmb, c);
        snprintf(nmb, sizeof nmb, "enc%d.%d.bn2.b", s, i);   addp(nmb, c);
      }
      snprintf(nmb, sizeof nmb, "up%d.w", s);    addp(nmb, (size_t)8 * CH[6 - s] * dec);
      snprintf(nmb, sizeof nmb, "up%d.bn.g", s); addp(nmb, dec);
      snprintf(nmb, sizeof nmb, "up%d.bn.b", s); addp(nmb, dec);
      for (int i = 0; i < LY[7 - s]; ++i) {
        int cin = (i == 0) ? dec + enc : dec;
        snprintf(nmb, sizeof nmb, "dec%d.%d.conv1.w", s, i); addp(nmb, (size_t)27 * cin * dec);
        snprintf(nmb, sizeof nmb, "dec%d.%d.bn1.g", s, i);   addp(nmb, dec);
        snprintf(nmb, sizeof nmb, "dec%d.%d.bn1.b", s, i);   addp(nmb, dec);
        snprintf(nmb, sizeof nmb, "dec%d.%d.conv2.w", s, i); addp(nmb, (size_t)27 * dec * dec);
        snprintf(nmb, sizeof nmb, "dec%d.%d.bn2.g", s, i);   addp(nmb, dec);
        snprintf(nmb, sizeof nmb, "dec%d.%d.bn2.b", s, i);   addp(nmb, dec);
        if (cin != dec) {
          snprintf(nmb, sizeof nmb, "dec%d.%d.proj.w", s, i);    addp(nmb, (size_t)cin * dec);
          snprintf(nmb, sizeof nmb, "dec%d.%d.proj.bn.g", s, i); addp(nmb, dec);
          snprintf(nmb, sizeof nmb, "dec%d.%d.proj.bn.b", s, i); addp(nmb, dec);
        }
      }
      enc = CH[s]; dec = CH[6 - s];
    }
  }
  addp("final.w", (size_t)96 * 13); addp("final.b", 13);

  if (n_in == 20) {               // params = one flat buffer, insertion order
    const float* base = (const float*)d_in[19];
    size_t o = 0;
    for (int i = 0; i < np; ++i) { tab[i].p = base + o; o += tab[i].cnt; }
  } else {                        // params = leaves in sorted-key (pytree) order
    int order[220];
    for (int i = 0; i < np; ++i) order[i] = i;
    std::sort(order, order + np, [&](int a, int b) { return strcmp(tab[a].name, tab[b].name) < 0; });
    int navail = n_in - 19;
    for (int j = 0; j < np && j < navail; ++j) tab[order[j]].p = (const float*)d_in[19 + j];
  }
  auto Pn = [&](const char* s)->const float* {
    for (int i = 0; i < np; ++i) if (!strcmp(tab[i].name, s)) return tab[i].p;
    return nullptr;
  };
  auto G1 = [&](const char* fmt, int a)->const float* {
    char s[64]; snprintf(s, sizeof s, fmt, a); return Pn(s);
  };
  auto G2 = [&](const char* fmt, int a, int b)->const float* {
    char s[64]; snprintf(s, sizeof s, fmt, a, b); return Pn(s);
  };

  // ---- workspace bump allocator ----
  char* wsb = (char*)d_ws; size_t wso = 0;
  auto walloc = [&](size_t bytes)->void* {
    size_t a = (wso + 255) & ~(size_t)255; wso = a + bytes; return (void*)(wsb + a);
  };

  // ---- launch helpers ----
  float* stats = (float*)walloc(512 * sizeof(float));
  auto sconv = [&](const _Float16* A16, const int* idx, const _Float16* W16, float* o,
                   int Nout, int K, int CinP, int Cout, int padRow) {
    int mt = (Nout + 127) / 128, nt = (Cout + 63) / 64;
    k_sconv<<<mt * nt, 256, 0, stream>>>(A16, idx, W16, o, Nout, K, CinP, Cout, padRow);
  };
  auto bnact = [&](float* t, const float* g, const float* b, const float* res,
                   float* d32, int s32, int o32, _Float16* d16, int s16, int o16,
                   int N, int C, int relu) {
    k_stats<<<C / 32, 256, 0, stream>>>(t, stats, N, C);
    size_t tot = (size_t)N * C;
    k_bn_act<<<(unsigned)((tot + 255) / 256), 256, 0, stream>>>(t, stats, g, b, res,
                                                                d32, s32, o32, d16, s16, o16, N, C, relu);
  };
  auto castW = [&](const float* w, int K, int Cin, int CinP, int Cout)->_Float16* {
    _Float16* d = (_Float16*)walloc((size_t)K * CinP * Cout * sizeof(_Float16));
    size_t tot = (size_t)K * CinP * Cout;
    k_castw<<<(unsigned)((tot + 255) / 256), 256, 0, stream>>>(w, d, K, Cin, CinP, Cout);
    return d;
  };

  // ---- channel bookkeeping ----
  const int hC[5]    = {32, 32, 64, 128, 256};   // encoder output channels / level
  const int dC[4]    = {96, 96, 128, 256};       // decoder output channels / stage
  const int upIn[4]  = {96, 128, 256, 256};
  const int skipC[4] = {32, 32, 64, 128};
  int catC[4]; for (int s = 0; s < 4; ++s) catC[s] = dC[s] + skipC[s];

  // ---- feature buffers ----
  float* h32[5]; _Float16* h16[5];
  for (int l = 0; l < 5; ++l) {
    h32[l] = (float*)walloc((size_t)Nl[l] * hC[l] * sizeof(float));
    h16[l] = (_Float16*)walloc((size_t)(Nl[l] + 1) * hC[l] * sizeof(_Float16));
    hipMemsetAsync(h16[l] + (size_t)Nl[l] * hC[l], 0, hC[l] * sizeof(_Float16), stream); // pad row
  }
  _Float16* x16 = (_Float16*)walloc((size_t)(Nl[0] + 1) * 32 * sizeof(_Float16));
  hipMemsetAsync(x16, 0, (size_t)(Nl[0] + 1) * 32 * sizeof(_Float16), stream);           // pad cols + row
  {
    size_t tot = (size_t)Nl[0] * 4;
    k_pack<<<(unsigned)((tot + 255) / 256), 256, 0, stream>>>(x, 4, nullptr, 0, 0, x16, 32, 0, Nl[0], 4);
  }
  float* cat32[4]; _Float16* cat16[4]; float* dd32[4]; _Float16* dd16[4];
  for (int s = 0; s < 4; ++s) {
    cat32[s] = (float*)walloc((size_t)Nl[s] * catC[s] * sizeof(float));
    cat16[s] = (_Float16*)walloc((size_t)(Nl[s] + 1) * catC[s] * sizeof(_Float16));
    hipMemsetAsync(cat16[s] + (size_t)Nl[s] * catC[s], 0, catC[s] * sizeof(_Float16), stream);
    dd32[s] = (float*)walloc((size_t)Nl[s] * dC[s] * sizeof(float));
    dd16[s] = (_Float16*)walloc((size_t)(Nl[s] + 1) * dC[s] * sizeof(_Float16));
    hipMemsetAsync(dd16[s] + (size_t)Nl[s] * dC[s], 0, dC[s] * sizeof(_Float16), stream);
  }
  size_t maxNC = (size_t)Nl[0] * 32;
  for (int s = 0; s < 4; ++s) {
    maxNC = std::max(maxNC, (size_t)Nl[s + 1] * CH[s]);
    maxNC = std::max(maxNC, (size_t)Nl[s] * dC[s]);
  }
  float* t1 = (float*)walloc(maxNC * sizeof(float));
  float* rp = (float*)walloc(maxNC * sizeof(float));
  size_t o16n = 0;
  for (int s = 0; s < 4; ++s) {
    o16n = std::max(o16n, (size_t)(Nl[s + 1] + 1) * CH[s]);
    o16n = std::max(o16n, (size_t)(Nl[s] + 1) * dC[s]);
  }
  _Float16* o16 = (_Float16*)walloc(o16n * sizeof(_Float16));
  int* invIdx = (int*)walloc((size_t)Nl[0] * 8 * sizeof(int));

  // ---- weight f16 conversion ----
  _Float16* Wci = castW(Pn("conv_input.w"), 125, 4, 32, 32);
  _Float16* Wdown[4]; _Float16* Wenc[4][6][2]; _Float16* Wup[4]; _Float16* Wdec[4][2][2]; _Float16* Wproj[4];
  for (int s = 0; s < 4; ++s) {
    Wdown[s] = castW(G1("down%d.w", s), 8, hC[s], hC[s], CH[s]);
    for (int i = 0; i < LY[s]; ++i) {
      Wenc[s][i][0] = castW(G2("enc%d.%d.conv1.w", s, i), 27, CH[s], CH[s], CH[s]);
      Wenc[s][i][1] = castW(G2("enc%d.%d.conv2.w", s, i), 27, CH[s], CH[s], CH[s]);
    }
    Wup[s] = castW(G1("up%d.w", s), 8, upIn[s], upIn[s], dC[s]);
    Wdec[s][0][0] = castW(G2("dec%d.%d.conv1.w", s, 0), 27, catC[s], catC[s], dC[s]);
    Wdec[s][0][1] = castW(G2("dec%d.%d.conv2.w", s, 0), 27, dC[s], dC[s], dC[s]);
    Wdec[s][1][0] = castW(G2("dec%d.%d.conv1.w", s, 1), 27, dC[s], dC[s], dC[s]);
    Wdec[s][1][1] = castW(G2("dec%d.%d.conv2.w", s, 1), 27, dC[s], dC[s], dC[s]);
    Wproj[s] = castW(G2("dec%d.%d.proj.w", s, 0), 1, catC[s], catC[s], dC[s]);
  }

  // ================= encoder =================
  sconv(x16, nb5, Wci, t1, Nl[0], 125, 32, 32, Nl[0]);
  bnact(t1, Pn("conv_input.bn.g"), Pn("conv_input.bn.b"), nullptr,
        h32[0], 32, 0, h16[0], 32, 0, Nl[0], 32, 1);

  for (int s = 0; s < 4; ++s) {
    int lvl = s + 1, C = CH[s], N = Nl[lvl];
    sconv(h16[s], dwn[s], Wdown[s], t1, N, 8, hC[s], C, Nl[s]);
    bnact(t1, G1("down%d.bn.g", s), G1("down%d.bn.b", s), nullptr,
          h32[lvl], C, 0, h16[lvl], C, 0, N, C, 1);
    for (int i = 0; i < LY[s]; ++i) {
      sconv(h16[lvl], nb3[lvl], Wenc[s][i][0], t1, N, 27, C, C, N);
      hipMemsetAsync(o16 + (size_t)N * C, 0, C * sizeof(_Float16), stream);
      bnact(t1, G2("enc%d.%d.bn1.g", s, i), G2("enc%d.%d.bn1.b", s, i), nullptr,
            nullptr, 0, 0, o16, C, 0, N, C, 1);
      sconv(o16, nb3[lvl], Wenc[s][i][1], t1, N, 27, C, C, N);
      bnact(t1, G2("enc%d.%d.bn2.g", s, i), G2("enc%d.%d.bn2.b", s, i), h32[lvl],
            h32[lvl], C, 0, h16[lvl], C, 0, N, C, 1);
    }
  }

  // ================= decoder =================
  const _Float16* cur16 = h16[4];
  for (int s = 3; s >= 0; --s) {
    int Nf = Nl[s], Nc = Nl[s + 1];
    k_invidx<<<(Nf + 255) / 256, 256, 0, stream>>>(par[s], off[s], invIdx, Nf, Nc);
    // up conv -> BN+ReLU directly into concat region cols [0, dC)
    sconv(cur16, invIdx, Wup[s], t1, Nf, 8, upIn[s], dC[s], Nc);
    bnact(t1, G1("up%d.bn.g", s), G1("up%d.bn.b", s), nullptr,
          cat32[s], catC[s], 0, cat16[s], catC[s], 0, Nf, dC[s], 1);
    // skip into cols [dC, catC)
    {
      size_t tot = (size_t)Nf * skipC[s];
      k_pack<<<(unsigned)((tot + 255) / 256), 256, 0, stream>>>(
          h32[s], skipC[s], cat32[s], catC[s], dC[s], cat16[s], catC[s], dC[s], Nf, skipC[s]);
    }
    for (int i = 0; i < 2; ++i) {
      const _Float16* fin = (i == 0) ? cat16[s] : dd16[s];
      int cin = (i == 0) ? catC[s] : dC[s];
      const float* res;
      if (i == 0) {
        sconv(cat16[s], nullptr, Wproj[s], rp, Nf, 1, catC[s], dC[s], Nf);      // proj GEMM -> rp raw
        k_stats<<<dC[s] / 32, 256, 0, stream>>>(rp, stats, Nf, dC[s]);
        size_t tot = (size_t)Nf * dC[s];
        k_bn_act<<<(unsigned)((tot + 255) / 256), 256, 0, stream>>>(
            rp, stats, G2("dec%d.%d.proj.bn.g", s, 0), G2("dec%d.%d.proj.bn.b", s, 0), nullptr,
            rp, dC[s], 0, nullptr, 0, 0, Nf, dC[s], 0);                          // BN in-place, no relu
        res = rp;
      } else {
        res = dd32[s];
      }
      sconv(fin, nb3[s], Wdec[s][i][0], t1, Nf, 27, cin, dC[s], Nf);
      hipMemsetAsync(o16 + (size_t)Nf * dC[s], 0, dC[s] * sizeof(_Float16), stream);
      bnact(t1, G2("dec%d.%d.bn1.g", s, i), G2("dec%d.%d.bn1.b", s, i), nullptr,
            nullptr, 0, 0, o16, dC[s], 0, Nf, dC[s], 1);
      sconv(o16, nb3[s], Wdec[s][i][1], t1, Nf, 27, dC[s], dC[s], Nf);
      bnact(t1, G2("dec%d.%d.bn2.g", s, i), G2("dec%d.%d.bn2.b", s, i), res,
            dd32[s], dC[s], 0, dd16[s], dC[s], 0, Nf, dC[s], 1);
    }
    cur16 = dd16[s];
  }

  // ================= final linear =================
  k_final<<<(Nl[0] + 127) / 128, 128, 0, stream>>>(dd32[0], Pn("final.w"), Pn("final.b"),
                                                   (float*)d_out, Nl[0]);
}
